// DrugGraphFEM_83889301225554
// MI455X (gfx1250) — compile-verified, matrix operand order checked by reading, not run on hardware
//
#include <hip/hip_runtime.h>
#include <hip/hip_bf16.h>
#include <stdint.h>

// ---------------- problem constants (match reference) ----------------
#define N_NODES 100000
#define N_EDGES 400000
#define ETOT    (N_EDGES + N_NODES)   // with self loops
#define GRAPHS  2048
#define F_IN    78
#define KPAD1   96                    // F_IN padded to multiple of 32
#define HEADS   4
#define CH      128
#define DHID    512
#define OUTD    256
#define BN_EPS  1e-5f

// ---------------- WMMA vector types (wave32, CDNA5) ----------------
typedef __bf16 bf16s;
typedef bf16s v16bf __attribute__((ext_vector_type(16)));
typedef float v8f  __attribute__((ext_vector_type(8)));

// order-preserving float->uint map so atomicMax(uint) == float max
__device__ __forceinline__ unsigned flipf(float f) {
    unsigned u = __float_as_uint(f);
    return (u & 0x80000000u) ? ~u : (u | 0x80000000u);
}
__device__ __forceinline__ float unflipf(unsigned v) {
    unsigned u = (v & 0x80000000u) ? (v ^ 0x80000000u) : ~v;
    return __uint_as_float(u);
}

// ---------------- generic fill ----------------
__global__ void k_fill_f32(float* __restrict__ p, float v, long long n) {
    long long i = (long long)blockIdx.x * blockDim.x + threadIdx.x;
    if (i < n) p[i] = v;
}

// ---------------- f32 -> bf16 row copy with K padding ----------------
__global__ void k_cvt_pad_rows(const float* __restrict__ X, __hip_bfloat16* __restrict__ Xb,
                               int rows, int Kin, int Kpad) {
    long long i = (long long)blockIdx.x * blockDim.x + threadIdx.x;
    if (i >= (long long)rows * Kpad) return;
    int r = (int)(i / Kpad), k = (int)(i % Kpad);
    Xb[i] = (k < Kin) ? __float2bfloat16(X[(size_t)r * Kin + k]) : __float2bfloat16(0.0f);
}

// W[K][Nout] f32 -> Wt[Nout][Kpad] bf16 (transposed, K zero-padded)
__global__ void k_cvt_transpose(const float* __restrict__ W, __hip_bfloat16* __restrict__ Wt,
                                int Kin, int Nout, int Kpad) {
    long long i = (long long)blockIdx.x * blockDim.x + threadIdx.x;
    if (i >= (long long)Nout * Kpad) return;
    int j = (int)(i / Kpad), k = (int)(i % Kpad);
    Wt[i] = (k < Kin) ? __float2bfloat16(W[(size_t)k * Nout + j]) : __float2bfloat16(0.0f);
}

// ---------------- WMMA fragment helpers ----------------
union Frag {
    uint4    q[2];
    unsigned u[8];
    v16bf    v;
};
struct FragSet { Frag a, b0, b1, b2, b3; };

__device__ __forceinline__ void load_frags(FragSet& s,
        const __hip_bfloat16* __restrict__ aRow,   // LDS row (ds_load path)
        const __hip_bfloat16* __restrict__ r0, const __hip_bfloat16* __restrict__ r1,
        const __hip_bfloat16* __restrict__ r2, const __hip_bfloat16* __restrict__ r3,
        int kb, int hlf) {
    // 16-bit A/B layout (ISA 7.12.2): per lane, dwords 0..3 cover K = kb+8*hlf .. +7,
    // dwords 4..7 cover K = kb+16+8*hlf .. +7  -> two 16-byte loads each.
    const int e0 = kb + 8 * hlf;
    const int e1 = e0 + 16;
    s.a.q[0]  = *(const uint4*)(aRow + e0);
    s.a.q[1]  = *(const uint4*)(aRow + e1);
    s.b0.q[0] = *(const uint4*)(r0 + e0);
    s.b0.q[1] = *(const uint4*)(r0 + e1);
    s.b1.q[0] = *(const uint4*)(r1 + e0);
    s.b1.q[1] = *(const uint4*)(r1 + e1);
    s.b2.q[0] = *(const uint4*)(r2 + e0);
    s.b2.q[1] = *(const uint4*)(r2 + e1);
    s.b3.q[0] = *(const uint4*)(r3 + e0);
    s.b3.q[1] = *(const uint4*)(r3 + e1);
}

__device__ __forceinline__ void mma_set(const FragSet& s, v8f& c0, v8f& c1, v8f& c2, v8f& c3) {
    c0 = __builtin_amdgcn_wmma_f32_16x16x32_bf16(false, s.a.v, false, s.b0.v, (short)0, c0, false, false);
    c1 = __builtin_amdgcn_wmma_f32_16x16x32_bf16(false, s.a.v, false, s.b1.v, (short)0, c1, false, false);
    c2 = __builtin_amdgcn_wmma_f32_16x16x32_bf16(false, s.a.v, false, s.b2.v, (short)0, c2, false, false);
    c3 = __builtin_amdgcn_wmma_f32_16x16x32_bf16(false, s.a.v, false, s.b3.v, (short)0, c3, false, false);
}

// ---------------- bf16 WMMA GEMM: Out[M][NTN*64] = X[M][K] * Wt[NTN*64][K]^T ----------------
// Compile-time K/NTN: chunk loop fully unrolls, double-buffer guards fold away.
// Block = 8 waves; 8/NTN row strips of 16. A tile (contiguous [rows_pb x K] bf16)
// staged into LDS via async-to-LDS DMA (ASYNCcnt); B rows stream from global (L2-resident).
template <int K, int NTN>
__global__ void k_gemm_bf16_wmma(const __hip_bfloat16* __restrict__ X,
                                 const __hip_bfloat16* __restrict__ Wt,
                                 float* __restrict__ Out,
                                 const float* __restrict__ bias,
                                 int M) {
    constexpr int STRIPS  = 8 / NTN;
    constexpr int ROWS_PB = STRIPS * 16;
    constexpr int NOUT    = NTN * 64;
    constexpr int NCH     = K / 32;

    __shared__ __align__(16) __hip_bfloat16 lA[ROWS_PB * K];

    const int warp  = threadIdx.x >> 5;
    const int lane  = threadIdx.x & 31;
    const int tn    = warp % NTN;
    const int strip = warp / NTN;
    const int row0  = blockIdx.x * ROWS_PB;     // grid sized exactly: no OOB blocks
    const int hlf   = lane >> 4;
    const int l16   = lane & 15;

    // ---- stage A tile: one contiguous global region -> LDS, async DMA ----
    {
        constexpr unsigned nbytes = (unsigned)ROWS_PB * (unsigned)K * 2u;
        const unsigned long long g0 =
            (unsigned long long)(const char*)(X + (size_t)row0 * K);
        const unsigned l0 = (unsigned)(size_t)(void*)&lA[0];  // low 32 bits = LDS byte addr
        for (unsigned t = (unsigned)threadIdx.x * 16u; t < nbytes; t += 256u * 16u) {
            unsigned            la = l0 + t;
            unsigned long long  ga = g0 + t;
            asm volatile("global_load_async_to_lds_b128 %0, %1, off"
                         :: "v"(la), "v"(ga) : "memory");
        }
        asm volatile("s_wait_asynccnt 0x0" ::: "memory");
        __syncthreads();
    }

    const __hip_bfloat16* aRow = lA + (size_t)(strip * 16 + l16) * K;
    const __hip_bfloat16* r0 = Wt + (size_t)(tn * 64 +  0 + l16) * K;
    const __hip_bfloat16* r1 = Wt + (size_t)(tn * 64 + 16 + l16) * K;
    const __hip_bfloat16* r2 = Wt + (size_t)(tn * 64 + 32 + l16) * K;
    const __hip_bfloat16* r3 = Wt + (size_t)(tn * 64 + 48 + l16) * K;

    v8f acc0 = {}, acc1 = {}, acc2 = {}, acc3 = {};

    // ---- double-buffered K loop (fully unrolled; guards are compile-time) ----
    FragSet s[2];
    load_frags(s[0], aRow, r0, r1, r2, r3, 0, hlf);
#pragma unroll
    for (int ch = 0; ch < NCH; ++ch) {
        if (ch + 1 < NCH) load_frags(s[(ch + 1) & 1], aRow, r0, r1, r2, r3, (ch + 1) * 32, hlf);
        mma_set(s[ch & 1], acc0, acc1, acc2, acc3);
    }

    // ---- store: lane -> col l16; VGPR r -> row r + 8*hlf ----
#pragma unroll
    for (int j = 0; j < 4; ++j) {
        v8f acc = (j == 0) ? acc0 : (j == 1) ? acc1 : (j == 2) ? acc2 : acc3;
        const int col = tn * 64 + j * 16 + l16;
        const float bj = bias ? bias[col] : 0.0f;
#pragma unroll
        for (int r = 0; r < 8; ++r) {
            const int row = row0 + strip * 16 + r + 8 * hlf;
            Out[(size_t)row * NOUT + col] = acc[r] + bj;
        }
    }
}

// ---------------- attention scores: e_s/e_d[n][h] = <h[n,h,:], a_src/a_dst[h,:]> ----------------
__global__ void k_scores(const float* __restrict__ Hh, const float* __restrict__ as_,
                         const float* __restrict__ ad_, float* __restrict__ es,
                         float* __restrict__ ed_, unsigned* __restrict__ nmax,
                         float* __restrict__ nsum, int n) {
    int w = (blockIdx.x * blockDim.x + threadIdx.x) >> 5;
    int lane = threadIdx.x & 31;
    if (w >= n * HEADS) return;
    int node = w >> 2, head = w & 3;
    const float* hrow = Hh + (size_t)node * DHID + head * CH;
    const float* av = as_ + head * CH;
    const float* dv = ad_ + head * CH;
    float s = 0.f, d = 0.f;
    for (int c = lane; c < CH; c += 32) { float hv = hrow[c]; s += hv * av[c]; d += hv * dv[c]; }
    for (int o = 16; o; o >>= 1) { s += __shfl_xor(s, o, 32); d += __shfl_xor(d, o, 32); }
    if (lane == 0) {
        es[w] = s; ed_[w] = d;
        nmax[w] = flipf(-3.0e38f);   // init segment-max
        nsum[w] = 0.f;               // init segment-sum
    }
}

__device__ __forceinline__ void edge_endpoints(const int* ei, int e, int& src, int& dst) {
    if (e < N_EDGES) { src = ei[e]; dst = ei[N_EDGES + e]; }
    else             { src = dst = e - N_EDGES; }   // self loops appended
}

// ---------------- segment max over dst ----------------
__global__ void k_edge_max(const int* __restrict__ ei, const float* __restrict__ es,
                           const float* __restrict__ ed_, unsigned* __restrict__ nmax) {
    int idx = blockIdx.x * blockDim.x + threadIdx.x;
    if (idx >= ETOT * HEADS) return;
    int e = idx >> 2, h = idx & 3;
    int src, dst; edge_endpoints(ei, e, src, dst);
    float a = es[src * HEADS + h] + ed_[dst * HEADS + h];
    a = (a >= 0.f) ? a : 0.2f * a;
    atomicMax(&nmax[dst * HEADS + h], flipf(a));
}

// ---------------- segment sum of exp ----------------
__global__ void k_edge_sum(const int* __restrict__ ei, const float* __restrict__ es,
                           const float* __restrict__ ed_, const unsigned* __restrict__ nmax,
                           float* __restrict__ nsum) {
    int idx = blockIdx.x * blockDim.x + threadIdx.x;
    if (idx >= ETOT * HEADS) return;
    int e = idx >> 2, h = idx & 3;
    int src, dst; edge_endpoints(ei, e, src, dst);
    float a = es[src * HEADS + h] + ed_[dst * HEADS + h];
    a = (a >= 0.f) ? a : 0.2f * a;
    float m = unflipf(nmax[dst * HEADS + h]);
    atomicAdd(&nsum[dst * HEADS + h], __expf(a - m));
}

// ---------------- per-(edge,head) softmax coefficient (computed ONCE, not per channel) ----------------
__global__ void k_edge_coef(const int* __restrict__ ei, const float* __restrict__ es,
                            const float* __restrict__ ed_, const unsigned* __restrict__ nmax,
                            const float* __restrict__ nsum, float* __restrict__ coefs) {
    int idx = blockIdx.x * blockDim.x + threadIdx.x;
    if (idx >= ETOT * HEADS) return;
    int e = idx >> 2, h = idx & 3;
    int src, dst; edge_endpoints(ei, e, src, dst);
    float a = es[src * HEADS + h] + ed_[dst * HEADS + h];
    a = (a >= 0.f) ? a : 0.2f * a;
    float m = unflipf(nmax[dst * HEADS + h]);
    coefs[idx] = __expf(a - m) / nsum[dst * HEADS + h];
}

// ---------------- weighted message aggregation (pure streaming + f32 atomics) ----------------
__global__ void k_edge_msg(const int* __restrict__ ei, const float* __restrict__ Hh,
                           const float* __restrict__ coefs, float* __restrict__ agg) {
    long long idx = (long long)blockIdx.x * blockDim.x + threadIdx.x;
    if (idx >= (long long)ETOT * DHID) return;
    int e = (int)(idx >> 9);
    int c = (int)(idx & 511);
    int h = c >> 7;
    int src, dst; edge_endpoints(ei, e, src, dst);
    float coef = coefs[e * HEADS + h];
    atomicAdd(&agg[(size_t)dst * DHID + c], Hh[(size_t)src * DHID + c] * coef);
}

// ---------------- +bias, leaky(0.01); keep f32, emit bf16 copy for next GEMM ----------------
__global__ void k_bias_act(float* __restrict__ agg, const float* __restrict__ b,
                           __hip_bfloat16* __restrict__ hb, long long total, int D) {
    long long i = (long long)blockIdx.x * blockDim.x + threadIdx.x;
    if (i >= total) return;
    float v = agg[i] + b[(int)(i % D)];
    v = (v >= 0.f) ? v : 0.01f * v;
    agg[i] = v;
    hb[i] = __float2bfloat16(v);
}

// ---------------- global mean pool ----------------
__global__ void k_pool_add(const float* __restrict__ hact, const int* __restrict__ batch,
                           float* __restrict__ sums) {
    long long i = (long long)blockIdx.x * blockDim.x + threadIdx.x;
    if (i >= (long long)N_NODES * DHID) return;
    int n = (int)(i >> 9), c = (int)(i & 511);
    atomicAdd(&sums[(size_t)batch[n] * DHID + c], hact[i]);
}
__global__ void k_pool_cnt(const int* __restrict__ batch, float* __restrict__ cnt) {
    int n = blockIdx.x * blockDim.x + threadIdx.x;
    if (n < N_NODES) atomicAdd(&cnt[batch[n]], 1.0f);
}
__global__ void k_pool_div(const float* __restrict__ sums, const float* __restrict__ cnt,
                           __hip_bfloat16* __restrict__ pb) {
    int i = blockIdx.x * blockDim.x + threadIdx.x;
    if (i >= GRAPHS * DHID) return;
    int g = i >> 9;
    pb[i] = __float2bfloat16(sums[i] / fmaxf(cnt[g], 1.0f));
}

// ---------------- fc1 bias + eval-BN + leaky, emit bf16 ----------------
__global__ void k_bn_act(const float* __restrict__ Z, const float* __restrict__ fc1b,
                         const float* __restrict__ g, const float* __restrict__ bb,
                         const float* __restrict__ m, const float* __restrict__ v,
                         __hip_bfloat16* __restrict__ zb) {
    int i = blockIdx.x * blockDim.x + threadIdx.x;
    if (i >= GRAPHS * OUTD) return;
    int c = i & (OUTD - 1);
    float z = Z[i] + fc1b[c];
    z = (z - m[c]) * rsqrtf(v[c] + BN_EPS) * g[c] + bb[c];
    z = (z >= 0.f) ? z : 0.01f * z;
    zb[i] = __float2bfloat16(z);
}

// ---------------- host-side launcher ----------------
static inline unsigned nblk(long long n, int tb) { return (unsigned)((n + tb - 1) / tb); }

extern "C" void kernel_launch(void* const* d_in, const int* in_sizes, int n_in,
                              void* d_out, int out_size, void* d_ws, size_t ws_size,
                              hipStream_t stream) {
    const float* x    = (const float*)d_in[0];
    const int*   ei   = (const int*)d_in[1];
    const int*   bat  = (const int*)d_in[2];
    const float* W1   = (const float*)d_in[3];
    const float* as1  = (const float*)d_in[4];
    const float* ad1  = (const float*)d_in[5];
    const float* b1   = (const float*)d_in[6];
    const float* W2   = (const float*)d_in[7];
    const float* as2  = (const float*)d_in[8];
    const float* ad2  = (const float*)d_in[9];
    const float* b2   = (const float*)d_in[10];
    const float* fc1w = (const float*)d_in[11];
    const float* fc1b = (const float*)d_in[12];
    const float* bng  = (const float*)d_in[13];
    const float* bnb  = (const float*)d_in[14];
    const float* bnm  = (const float*)d_in[15];
    const float* bnv  = (const float*)d_in[16];
    const float* fc2w = (const float*)d_in[17];
    const float* fc2b = (const float*)d_in[18];
    float* out = (float*)d_out;

    // workspace carve-up (deterministic)
    char* ws = (char*)d_ws;
    size_t off = 0;
    auto take = [&](size_t bytes) -> void* {
        off = (off + 255) & ~(size_t)255;
        void* p = ws + off; off += bytes; return p;
    };
    __hip_bfloat16* xb   = (__hip_bfloat16*)take((size_t)N_NODES * KPAD1 * 2);
    __hip_bfloat16* W1t  = (__hip_bfloat16*)take((size_t)DHID * KPAD1 * 2);
    __hip_bfloat16* W2t  = (__hip_bfloat16*)take((size_t)DHID * DHID * 2);
    __hip_bfloat16* F1t  = (__hip_bfloat16*)take((size_t)OUTD * DHID * 2);
    __hip_bfloat16* F2t  = (__hip_bfloat16*)take((size_t)OUTD * OUTD * 2);
    float*          Hh   = (float*)take((size_t)N_NODES * DHID * 4);
    float*          AGG  = (float*)take((size_t)N_NODES * DHID * 4);
    __hip_bfloat16* HB   = (__hip_bfloat16*)take((size_t)N_NODES * DHID * 2);
    float*          es   = (float*)take((size_t)N_NODES * HEADS * 4);
    float*          edv  = (float*)take((size_t)N_NODES * HEADS * 4);
    unsigned*       nmx  = (unsigned*)take((size_t)N_NODES * HEADS * 4);
    float*          nsm  = (float*)take((size_t)N_NODES * HEADS * 4);
    float*          cofs = (float*)take((size_t)ETOT * HEADS * 4);
    float*          sums = (float*)take((size_t)GRAPHS * DHID * 4);
    float*          cnt  = (float*)take((size_t)GRAPHS * 4);
    __hip_bfloat16* pb   = (__hip_bfloat16*)take((size_t)GRAPHS * DHID * 2);
    float*          Z1   = (float*)take((size_t)GRAPHS * OUTD * 4);
    __hip_bfloat16* zb   = (__hip_bfloat16*)take((size_t)GRAPHS * OUTD * 2);
    if (off > ws_size) return;   // insufficient workspace

    const int TB = 256;
    const long long ND = (long long)N_NODES * DHID;
    const long long ED = (long long)ETOT * DHID;

    // weight / input conversions (bf16, transposed, K padded to multiple of 32)
    k_cvt_pad_rows<<<nblk((long long)N_NODES * KPAD1, TB), TB, 0, stream>>>(x, xb, N_NODES, F_IN, KPAD1);
    k_cvt_transpose<<<nblk((long long)DHID * KPAD1, TB), TB, 0, stream>>>(W1, W1t, F_IN, DHID, KPAD1);
    k_cvt_transpose<<<nblk((long long)DHID * DHID, TB), TB, 0, stream>>>(W2, W2t, DHID, DHID, DHID);
    k_cvt_transpose<<<nblk((long long)OUTD * DHID, TB), TB, 0, stream>>>(fc1w, F1t, DHID, OUTD, DHID);
    k_cvt_transpose<<<nblk((long long)OUTD * OUTD, TB), TB, 0, stream>>>(fc2w, F2t, OUTD, OUTD, OUTD);

    // ---- GAT layer 1 ----
    {
        k_gemm_bf16_wmma<KPAD1, 8><<<N_NODES / 16, TB, 0, stream>>>(xb, W1t, Hh, nullptr, N_NODES);
        k_scores<<<nblk((long long)N_NODES * HEADS * 32, TB), TB, 0, stream>>>(Hh, as1, ad1, es, edv, nmx, nsm, N_NODES);
        k_fill_f32<<<nblk(ND, TB), TB, 0, stream>>>(AGG, 0.f, ND);
        k_edge_max<<<nblk((long long)ETOT * HEADS, TB), TB, 0, stream>>>(ei, es, edv, nmx);
        k_edge_sum<<<nblk((long long)ETOT * HEADS, TB), TB, 0, stream>>>(ei, es, edv, nmx, nsm);
        k_edge_coef<<<nblk((long long)ETOT * HEADS, TB), TB, 0, stream>>>(ei, es, edv, nmx, nsm, cofs);
        k_edge_msg<<<nblk(ED, TB), TB, 0, stream>>>(ei, Hh, cofs, AGG);
        k_bias_act<<<nblk(ND, TB), TB, 0, stream>>>(AGG, b1, HB, ND, DHID);
    }
    // ---- GAT layer 2 ----
    {
        k_gemm_bf16_wmma<DHID, 8><<<N_NODES / 16, TB, 0, stream>>>(HB, W2t, Hh, nullptr, N_NODES);
        k_scores<<<nblk((long long)N_NODES * HEADS * 32, TB), TB, 0, stream>>>(Hh, as2, ad2, es, edv, nmx, nsm, N_NODES);
        k_fill_f32<<<nblk(ND, TB), TB, 0, stream>>>(AGG, 0.f, ND);
        k_edge_max<<<nblk((long long)ETOT * HEADS, TB), TB, 0, stream>>>(ei, es, edv, nmx);
        k_edge_sum<<<nblk((long long)ETOT * HEADS, TB), TB, 0, stream>>>(ei, es, edv, nmx, nsm);
        k_edge_coef<<<nblk((long long)ETOT * HEADS, TB), TB, 0, stream>>>(ei, es, edv, nmx, nsm, cofs);
        k_edge_msg<<<nblk(ED, TB), TB, 0, stream>>>(ei, Hh, cofs, AGG);
        k_bias_act<<<nblk(ND, TB), TB, 0, stream>>>(AGG, b2, HB, ND, DHID);
    }
    // ---- global mean pool ----
    k_fill_f32<<<nblk((long long)GRAPHS * DHID, TB), TB, 0, stream>>>(sums, 0.f, (long long)GRAPHS * DHID);
    k_fill_f32<<<nblk(GRAPHS, TB), TB, 0, stream>>>(cnt, 0.f, GRAPHS);
    k_pool_add<<<nblk(ND, TB), TB, 0, stream>>>(AGG, bat, sums);
    k_pool_cnt<<<nblk(N_NODES, TB), TB, 0, stream>>>(bat, cnt);
    k_pool_div<<<nblk((long long)GRAPHS * DHID, TB), TB, 0, stream>>>(sums, cnt, pb);

    // ---- fc head: Linear -> BN -> LeakyReLU -> Linear ----
    {
        // NTN = OUTD/64 = 4 -> 2 row strips per block -> 32 rows per block
        k_gemm_bf16_wmma<DHID, 4><<<GRAPHS / 32, TB, 0, stream>>>(pb, F1t, Z1, nullptr, GRAPHS);
        k_bn_act<<<nblk((long long)GRAPHS * OUTD, TB), TB, 0, stream>>>(Z1, fc1b, bng, bnb, bnm, bnv, zb);
        k_gemm_bf16_wmma<OUTD, 4><<<GRAPHS / 32, TB, 0, stream>>>(zb, F2t, out, fc2b, GRAPHS);
    }
}
